// DeepFM_83356725281234
// MI455X (gfx1250) — matrix operand order repaired and checked by reading, not compiled, not run on hardware
//
#include <hip/hip_runtime.h>
#include <hip/hip_bf16.h>
#include <math.h>

// ---------------- problem constants ----------------
#define BATCH      16384
#define EDIM       16
#define ONEHOT_N   26
#define MULTIHOT_N 2
#define LHOT       50
#define DENSE_N    13
#define NN_IN_REAL 461           // 28*16 + 13
#define NN_IN_PAD  480           // padded to multiple of 32 (WMMA K-step)
#define U0         64
#define U1         12
#define U1_PAD     16

// ---------------- WMMA vector types ----------------
typedef __attribute__((ext_vector_type(16))) __bf16       v16bf;
typedef __attribute__((ext_vector_type(8)))  float        v8f;
typedef __attribute__((ext_vector_type(4)))  unsigned int v4u;

union BfPack { v4u u[2]; v16bf b; };

// A-operand: two 16-byte chunks (16 consecutive K for this lane's row)
__device__ __forceinline__ v16bf load_a(const __bf16* p0, const __bf16* p1) {
    BfPack pk;
    pk.u[0] = *(const v4u*)p0;
    pk.u[1] = *(const v4u*)p1;
    return pk.b;
}
// B-operand: 32 contiguous bytes (16 consecutive K for this lane's column)
__device__ __forceinline__ v16bf load_b(const __bf16* p) {
    BfPack pk;
    pk.u[0] = *(const v4u*)p;
    pk.u[1] = *(const v4u*)(p + 8);
    return pk.b;
}

// ---------------- K0: weight convert/transpose/pad to bf16 ----------------
// WT0[n][k] = nn_w0[k][n]  (64 x 480 bf16, zero-padded k>=461)
// WT1[n][k] = nn_w1[k][n]  (16 x 64  bf16, zero-padded n>=12)
__global__ void k0_convert(const float* __restrict__ w0,
                           const float* __restrict__ w1,
                           __bf16* __restrict__ wt0,
                           __bf16* __restrict__ wt1) {
    int i = blockIdx.x * blockDim.x + threadIdx.x;
    const int n0 = U0 * NN_IN_PAD;
    if (i < n0) {
        int n = i / NN_IN_PAD, k = i % NN_IN_PAD;
        float v = (k < NN_IN_REAL) ? w0[k * U0 + n] : 0.0f;
        wt0[i] = (__bf16)v;
    } else if (i < n0 + U1_PAD * U0) {
        int j = i - n0;
        int n = j / U0, k = j % U0;
        float v = (n < U1) ? w1[k * U1 + n] : 0.0f;
        wt1[j] = (__bf16)v;
    }
}

// ---------------- K1: embedding gather + FM + build nn_inp ----------------
// 16 lanes per sample (lane = embedding dim), 64 samples per 256-thread block.
__global__ void __launch_bounds__(256)
k1_embed_fm(const float* __restrict__ dense,
            const int*   __restrict__ onehot,
            const int*   __restrict__ mh_ids,
            const float* __restrict__ mh_mask,
            const float* __restrict__ fm_w,
            const float* __restrict__ fm_emb,
            const float* __restrict__ concat_w,
            const float* __restrict__ concat_b,
            __bf16* __restrict__ nn_inp,
            float*  __restrict__ fmdot) {
    const int tid = threadIdx.x;
    const int e   = tid & 15;      // embedding dim handled by this lane
    const int grp = tid >> 4;      // 16 sample-groups per block

    for (int it = 0; it < 4; ++it) {
        const int b = blockIdx.x * 64 + it * 16 + grp;
        __bf16* nrow = nn_inp + (size_t)b * NN_IN_PAD;

        float s = 0.f, sq = 0.f, fm1 = 0.f;

        // ---- onehot fields: gather, accumulate FM stats, emit bf16 ----
        const int* oh = onehot + b * ONEHOT_N;
        for (int f = 0; f < ONEHOT_N; ++f) {
            const int id = oh[f];
            const float v = fm_emb[(size_t)id * EDIM + e];   // 64B coalesced/group
            s += v; sq += v * v;
            nrow[f * EDIM + e] = (__bf16)v;
            if (e == 0) fm1 += fm_w[id];                     // FM 1st order
        }

        // ---- multihot fields: masked mean over 50 slots ----
        for (int f = 0; f < MULTIHOT_N; ++f) {
            const int*   ids = mh_ids  + ((size_t)b * MULTIHOT_N + f) * LHOT;
            const float* mk  = mh_mask + ((size_t)b * MULTIHOT_N + f) * LHOT;
            float acc = 0.f, cnt = 0.f;
            for (int j = 0; j < LHOT; ++j) {
                const float m = mk[j];
                acc += fm_emb[(size_t)ids[j] * EDIM + e] * m;
                cnt += m;
            }
            const float v = acc / fmaxf(cnt, 1.0f);
            s += v; sq += v * v;
            nrow[(ONEHOT_N + f) * EDIM + e] = (__bf16)v;
        }

        // ---- dense features + zero padding (cols 448..479) ----
        for (int c = 448 + e; c < NN_IN_PAD; c += 16) {
            const int dc = c - 448;
            const float v = (dc < DENSE_N) ? dense[(size_t)b * DENSE_N + dc] : 0.0f;
            nrow[c] = (__bf16)v;
        }

        // ---- FM 2nd order dotted with concat_w, reduced over the 16 lanes ----
        float p = concat_w[1 + e] * 0.5f * (s * s - sq);
        p += __shfl_xor(p, 8, 16);
        p += __shfl_xor(p, 4, 16);
        p += __shfl_xor(p, 2, 16);
        p += __shfl_xor(p, 1, 16);
        if (e == 0)
            fmdot[b] = p + concat_w[0] * fm1 + concat_b[0];
    }
}

// ---------------- K2: fused MLP (WMMA) + final dot + sigmoid ----------------
// 256 threads = 8 waves; 128 samples per block. Wave w owns rows [w*16, w*16+16).
__global__ void __launch_bounds__(256)
k2_mlp(const __bf16* __restrict__ nn_inp,
       const __bf16* __restrict__ wt0,
       const __bf16* __restrict__ wt1,
       const float*  __restrict__ b0,
       const float*  __restrict__ b1,
       const float*  __restrict__ concat_w,
       const float*  __restrict__ fmdot,
       float* __restrict__ out) {
    __shared__ __bf16 sWT0[U0 * NN_IN_PAD];   // 61440 B
    __shared__ __bf16 sWT1[U1_PAD * U0];      //  2048 B
    __shared__ __bf16 sH0[128 * U0];          // 16384 B
    __shared__ float  sH1[128 * U1_PAD];      //  8192 B

    const int tid = threadIdx.x;

    // cooperative stage of weights into LDS (uint4 copies)
    {
        const v4u* g0 = (const v4u*)wt0;  v4u* l0 = (v4u*)sWT0;
        for (int i = tid; i < (U0 * NN_IN_PAD) / 8; i += 256) l0[i] = g0[i];
        const v4u* g1 = (const v4u*)wt1;  v4u* l1 = (v4u*)sWT1;
        for (int i = tid; i < (U1_PAD * U0) / 8; i += 256) l1[i] = g1[i];
    }
    __syncthreads();

    const int w  = tid >> 5;       // wave id 0..7
    const int l  = tid & 31;
    const int hi = l >> 4;         // half-wave select (ISA A/B lane layout)
    const int m  = l & 15;         // row (A) / column (B) within tile
    const int rowBase = blockIdx.x * 128;

    // ---- GEMM1: [16 x 480] x [480 x 64] per wave, K-step 32 ----
    v8f acc0 = {}, acc1 = {}, acc2 = {}, acc3 = {};
    const __bf16* arow = nn_inp + (size_t)(rowBase + w * 16 + m) * NN_IN_PAD;
    for (int ks = 0; ks < NN_IN_PAD / 32; ++ks) {
        const int ka = ks * 32;
        v16bf a  = load_a(arow + ka + hi * 16, arow + ka + hi * 16 + 8);
        v16bf q0 = load_b(sWT0 + ( 0 + m) * NN_IN_PAD + ka + hi * 16);
        v16bf q1 = load_b(sWT0 + (16 + m) * NN_IN_PAD + ka + hi * 16);
        v16bf q2 = load_b(sWT0 + (32 + m) * NN_IN_PAD + ka + hi * 16);
        v16bf q3 = load_b(sWT0 + (48 + m) * NN_IN_PAD + ka + hi * 16);
        acc0 = __builtin_amdgcn_wmma_f32_16x16x32_bf16(false, a, false, q0, (short)0, acc0, false, false);
        acc1 = __builtin_amdgcn_wmma_f32_16x16x32_bf16(false, a, false, q1, (short)0, acc1, false, false);
        acc2 = __builtin_amdgcn_wmma_f32_16x16x32_bf16(false, a, false, q2, (short)0, acc2, false, false);
        acc3 = __builtin_amdgcn_wmma_f32_16x16x32_bf16(false, a, false, q3, (short)0, acc3, false, false);
    }
    // bias + relu -> bf16 h0 in LDS (C layout: elem r -> row r + 8*hi, col m)
    {
        const int r0 = w * 16 + hi * 8;
        const float bb0 = b0[m], bb1 = b0[16 + m], bb2 = b0[32 + m], bb3 = b0[48 + m];
        for (int r = 0; r < 8; ++r) {
            sH0[(r0 + r) * U0 +      m] = (__bf16)fmaxf(acc0[r] + bb0, 0.f);
            sH0[(r0 + r) * U0 + 16 + m] = (__bf16)fmaxf(acc1[r] + bb1, 0.f);
            sH0[(r0 + r) * U0 + 32 + m] = (__bf16)fmaxf(acc2[r] + bb2, 0.f);
            sH0[(r0 + r) * U0 + 48 + m] = (__bf16)fmaxf(acc3[r] + bb3, 0.f);
        }
    }
    __syncthreads();

    // ---- GEMM2: [16 x 64] x [64 x 16(12)] per wave ----
    v8f acc = {};
    const __bf16* hrow = sH0 + (w * 16 + m) * U0;
    for (int ks = 0; ks < 2; ++ks) {
        const int ka = ks * 32;
        v16bf a  = load_a(hrow + ka + hi * 16, hrow + ka + hi * 16 + 8);
        v16bf bq = load_b(sWT1 + m * U0 + ka + hi * 16);
        acc = __builtin_amdgcn_wmma_f32_16x16x32_bf16(false, a, false, bq, (short)0, acc, false, false);
    }
    {
        const int r0 = w * 16 + hi * 8;
        const float bias = (m < U1) ? b1[m] : 0.0f;
        for (int r = 0; r < 8; ++r)
            sH1[(r0 + r) * U1_PAD + m] = fmaxf(acc[r] + bias, 0.f);
    }
    __syncthreads();

    // ---- final: fmdot + h1 . concat_w[17:29], sigmoid ----
    if (tid < 128) {
        const int b = rowBase + tid;
        float a = fmdot[b];
        for (int j = 0; j < U1; ++j)
            a += sH1[tid * U1_PAD + j] * concat_w[17 + j];
        out[b] = 1.0f / (1.0f + expf(-a));
    }
}

// ---------------- launch ----------------
extern "C" void kernel_launch(void* const* d_in, const int* in_sizes, int n_in,
                              void* d_out, int out_size, void* d_ws, size_t ws_size,
                              hipStream_t stream) {
    (void)in_sizes; (void)n_in; (void)out_size; (void)ws_size;
    const float* dense    = (const float*)d_in[0];
    const int*   onehot   = (const int*)  d_in[1];
    const int*   mh_ids   = (const int*)  d_in[2];
    const float* mh_mask  = (const float*)d_in[3];
    const float* fm_w     = (const float*)d_in[4];
    const float* fm_emb   = (const float*)d_in[5];
    const float* nn_w0    = (const float*)d_in[6];
    const float* nn_b0    = (const float*)d_in[7];
    const float* nn_w1    = (const float*)d_in[8];
    const float* nn_b1    = (const float*)d_in[9];
    const float* concat_w = (const float*)d_in[10];
    const float* concat_b = (const float*)d_in[11];
    float* out = (float*)d_out;

    // workspace layout (bytes, 256-aligned regions)
    char* ws = (char*)d_ws;
    __bf16* nn_inp = (__bf16*)(ws + 0);            // 16384*480*2 = 15,728,640
    __bf16* wt0    = (__bf16*)(ws + 15728640);     // 64*480*2    =     61,440
    __bf16* wt1    = (__bf16*)(ws + 15790080);     // 16*64*2     =      2,048
    float*  fmdot  = (float*) (ws + 15792128);     // 16384*4     =     65,536

    const int convN = U0 * NN_IN_PAD + U1_PAD * U0;    // 31744
    k0_convert<<<(convN + 255) / 256, 256, 0, stream>>>(nn_w0, nn_w1, wt0, wt1);

    k1_embed_fm<<<BATCH / 64, 256, 0, stream>>>(dense, onehot, mh_ids, mh_mask,
                                                fm_w, fm_emb, concat_w, concat_b,
                                                nn_inp, fmdot);

    k2_mlp<<<BATCH / 128, 256, 0, stream>>>(nn_inp, wt0, wt1, nn_b0, nn_b1,
                                            concat_w, fmdot, out);
}
